// SelfAttention_35390530519579
// MI455X (gfx1250) — matrix operand order repaired and checked by reading, not compiled, not run on hardware
//
#include <hip/hip_runtime.h>

#define B_ 4
#define T_ 2048
#define K_ 128
#define H_ 8
#define INNER 1024  // H*K

typedef __attribute__((ext_vector_type(16))) _Float16 v16h;
typedef __attribute__((ext_vector_type(8)))  float    f32x8;
typedef unsigned __attribute__((ext_vector_type(4))) u32x4;
typedef int __attribute__((ext_vector_type(4)))      i32x4;
typedef int __attribute__((ext_vector_type(8)))      i32x8;

#ifndef __has_builtin
#define __has_builtin(x) 0
#endif
#if __has_builtin(__builtin_amdgcn_tensor_load_to_lds) && \
    __has_builtin(__builtin_amdgcn_s_wait_tensorcnt)
#define USE_TDM 1
#else
#define USE_TDM 0
#endif

__device__ __forceinline__ f32x8 wmma_f16(v16h a, v16h b, f32x8 c) {
    // (neg_a, A, neg_b, B, c_mod, C, reuse_a, reuse_b)
    return __builtin_amdgcn_wmma_f32_16x16x32_f16(false, a, false, b, (short)0, c,
                                                  false, false);
}

// K-index inside a 16-bit 16x32 A fragment for element e (0..15), lane-half hl.
__device__ __forceinline__ int a_kidx(int e, int hl) {
    return e + (hl << 4);
}

#if USE_TDM
// TDM: DMA a [tile_h x tile_w] f16 tile (row-major, row stride row_stride
// elements) from global memory into LDS at lds_byte_addr (packed rows).
// 2-D descriptor per CDNA5 ISA.
__device__ __forceinline__ void tdm_load_tile(const _Float16* gsrc,
                                              unsigned lds_byte_addr,
                                              int tile_w, int tile_h,
                                              int row_stride) {
    unsigned long long ga = (unsigned long long)(size_t)gsrc;
    u32x4 g0;
    g0[0] = 1u;                                   // count=1, user descriptor
    g0[1] = lds_byte_addr;                        // lds_addr [63:32]
    g0[2] = (unsigned)(ga & 0xFFFFFFFFull);       // global_addr low
    g0[3] = (unsigned)((ga >> 32) & 0x01FFFFFFull) | (2u << 30);  // addr hi | type=2
    i32x8 g1;
    g1[0] = 0x00010000;                    // workgroup_mask=0, data_size=1 (2B)
    g1[1] = (int)((unsigned)tile_w << 16); // tensor_dim0 (bits 79:48) = tile_w
    g1[2] = (int)((unsigned)tile_h << 16); // tensor_dim1 (bits 111:80) = tile_h
    g1[3] = (int)((unsigned)tile_w << 16); // tile_dim0 (bits 127:112)
    g1[4] = tile_h;                        // tile_dim1, tile_dim2=0
    g1[5] = row_stride;                    // tensor_dim0_stride (bits 207:160)
    g1[6] = 0;
    g1[7] = 0;
    i32x4 g2 = {0, 0, 0, 0};               // 2-D: groups 2/3 unused
    i32x4 g3 = {0, 0, 0, 0};
#if defined(__clang_major__) && (__clang_major__ >= 23)
    i32x8 g4 = {0, 0, 0, 0, 0, 0, 0, 0};
    __builtin_amdgcn_tensor_load_to_lds(g0, g1, g2, g3, g4, 0);
#else
    __builtin_amdgcn_tensor_load_to_lds(g0, g1, g2, g3, 0);
#endif
}
#endif

// ---------------------------------------------------------------------------
// Prep: x (f32) -> Xh (f16), row-major [B*T][128].
// ---------------------------------------------------------------------------
__global__ void __launch_bounds__(256)
cvt_x_kernel(const float* __restrict__ X, _Float16* __restrict__ Xh) {
    int i = blockIdx.x * 256 + threadIdx.x;       // over element pairs
    float2 v = ((const float2*)X)[i];
    Xh[2 * i]     = (_Float16)v.x;
    Xh[2 * i + 1] = (_Float16)v.y;
}

// Prep: WT[n][k] = scale * W[k][n], f32 -> f16. W is [rows][cols].
__global__ void cvt_wT_kernel(const float* __restrict__ W, _Float16* __restrict__ WT,
                              int rows, int cols, float scale) {
    const int n = blockIdx.x;                     // output row = input column
    for (int k = threadIdx.x; k < rows; k += blockDim.x)
        WT[(size_t)n * rows + k] = (_Float16)(scale * W[(size_t)k * cols + n]);
}

// ---------------------------------------------------------------------------
// Fused QKV. Q,K written head-major [b][h][t][d]; V written head-TRANSPOSED
// [b][h][d][t] so attention can TDM-fetch [d][t] tiles for contiguous B-frags.
// Softmax / value scales folded into WqT / WvT.
// ---------------------------------------------------------------------------
__global__ void __launch_bounds__(32)
qkv_kernel(const _Float16* __restrict__ Xh,
           const _Float16* __restrict__ WqT, const _Float16* __restrict__ WkT,
           const _Float16* __restrict__ WvT,
           _Float16* __restrict__ Qo, _Float16* __restrict__ Ko,
           _Float16* __restrict__ Vt) {
    const int lane = threadIdx.x & 31;
    const int ln = lane & 15, hl = lane >> 4;
    const int Mbase = blockIdx.x * 16;
    const int Nbase = blockIdx.y * 16;
    const int n = Nbase + ln;

    v16h xa[4];
    {
        const _Float16* xrow = Xh + (size_t)(Mbase + ln) * K_;
#pragma unroll
        for (int ks = 0; ks < 4; ++ks) {
            union { v16h v; unsigned u[8]; } a;
#pragma unroll
            for (int g = 0; g < 8; ++g)
                a.u[g] = *(const unsigned*)&xrow[ks * 32 + a_kidx(2 * g, hl)];
            xa[ks] = a.v;
        }
    }

    f32x8 qacc = {}, kacc = {}, vacc = {};
#pragma unroll
    for (int ks = 0; ks < 4; ++ks) {
        const size_t boff = (size_t)n * K_ + ks * 32 + (hl << 4);
        qacc = wmma_f16(xa[ks], *(const v16h*)&WqT[boff], qacc);
        kacc = wmma_f16(xa[ks], *(const v16h*)&WkT[boff], kacc);
        vacc = wmma_f16(xa[ks], *(const v16h*)&WvT[boff], vacc);
    }

    const int h = n >> 7, d = n & 127;
#pragma unroll
    for (int r = 0; r < 8; ++r) {
        int m = Mbase + r + (hl << 3);
        int bb = m / T_, t = m - bb * T_;
        size_t hoff = ((size_t)bb * H_ + h) * T_ * K_;
        Qo[hoff + (size_t)t * K_ + d] = (_Float16)qacc[r];
        Ko[hoff + (size_t)t * K_ + d] = (_Float16)kacc[r];
        Vt[hoff + (size_t)d * T_ + t] = (_Float16)vacc[r];   // transposed
    }
}

// ---------------------------------------------------------------------------
// Flash attention. grid = (T/64, B*H), block = 128 (4 waves).
// K tiles [32 x 128] and V tiles [128 x 32] (strided) staged to LDS by the
// Tensor Data Mover, DOUBLE-BUFFERED: wave 0 issues tile it+1's DMA before
// computing tile it, then waits tensorcnt<=2 (in-order completion => current
// tile's two transfers are done). Fallback: synchronous b128 copies.
// ---------------------------------------------------------------------------
__global__ void __launch_bounds__(128)
attn_kernel(const _Float16* __restrict__ Qh, const _Float16* __restrict__ Kh,
            const _Float16* __restrict__ Vt, _Float16* __restrict__ Oh) {
    __shared__ __align__(16) _Float16 Ks[2][32 * K_];   // [s][d]
    __shared__ __align__(16) _Float16 Vs[2][K_ * 32];   // [d][s]
    __shared__ __align__(16) _Float16 Ps[4 * 16 * 32];

    const int tid = threadIdx.x;
    const int wave = tid >> 5;
    const int lane = tid & 31;
    const int ln = lane & 15, hl = lane >> 4;
    const int bh = blockIdx.y;
    const int qbase = blockIdx.x * 64 + wave * 16;

    const size_t headOff = (size_t)bh * T_ * K_;
    const _Float16* Qg = Qh + headOff;
    const _Float16* Kg = Kh + headOff;
    const _Float16* Vg = Vt + headOff;               // [d][t], row stride T_

    v16h qa[4];
    {
        const _Float16* qrow = Qg + (size_t)(qbase + ln) * K_;
#pragma unroll
        for (int ks = 0; ks < 4; ++ks) {
            union { v16h v; unsigned u[8]; } a;
#pragma unroll
            for (int g = 0; g < 8; ++g)
                a.u[g] = *(const unsigned*)&qrow[ks * 32 + a_kidx(2 * g, hl)];
            qa[ks] = a.v;
        }
    }

    f32x8 o[8];
#pragma unroll
    for (int dt = 0; dt < 8; ++dt) o[dt] = (f32x8){};
    float mrow[8], lrow[8];
#pragma unroll
    for (int r = 0; r < 8; ++r) { mrow[r] = -1e38f; lrow[r] = 0.0f; }

    _Float16* Pw = Ps + wave * (16 * 32);
    const int iters = 2 * blockIdx.x + 2;

#if USE_TDM
    unsigned ldsK[2], ldsV[2];
    ldsK[0] = (unsigned)(size_t)&Ks[0][0];
    ldsK[1] = (unsigned)(size_t)&Ks[1][0];
    ldsV[0] = (unsigned)(size_t)&Vs[0][0];
    ldsV[1] = (unsigned)(size_t)&Vs[1][0];
    if (wave == 0) {                                  // preload tile 0
        tdm_load_tile(Kg, ldsK[0], K_, 32, K_);
        tdm_load_tile(Vg, ldsV[0], 32, K_, T_);
    }
#endif

    for (int it = 0; it < iters; ++it) {
        const int s0 = it * 32;
        const int cur = it & 1;

#if USE_TDM
        if (wave == 0) {
            if (it + 1 < iters) {                     // prefetch next tile
                const int s1 = s0 + 32;
                tdm_load_tile(Kg + (size_t)s1 * K_, ldsK[cur ^ 1], K_, 32, K_);
                tdm_load_tile(Vg + s1,              ldsV[cur ^ 1], 32, K_, T_);
                __builtin_amdgcn_s_wait_tensorcnt(2);  // current tile done
            } else {
                __builtin_amdgcn_s_wait_tensorcnt(0);
            }
        }
        __syncthreads();
#else
        {
            const uint4* ksrc = (const uint4*)(Kg + (size_t)s0 * K_);
            uint4* kdst = (uint4*)&Ks[cur][0];
            uint4* vdst = (uint4*)&Vs[cur][0];
            for (int c = tid; c < 512; c += 128) {
                kdst[c] = ksrc[c];
                // V: 128 rows x 32 halves (64B = 4 uint4 per row), stride T_.
                int vr = c >> 2, vc = c & 3;
                vdst[c] = *(const uint4*)(Vg + (size_t)vr * T_ + s0 + vc * 8);
            }
        }
        __syncthreads();
#endif
        const _Float16* Kt = &Ks[cur][0];
        const _Float16* Vv = &Vs[cur][0];

        // S[16x32] = Q (16x128) * K^T
        f32x8 s0a = {}, s1a = {};
#pragma unroll
        for (int ks = 0; ks < 4; ++ks) {
            v16h b0 = *(const v16h*)&Kt[(size_t)ln * K_ + ks * 32 + (hl << 4)];
            v16h b1 = *(const v16h*)&Kt[(size_t)(ln + 16) * K_ + ks * 32 + (hl << 4)];
            s0a = wmma_f16(qa[ks], b0, s0a);
            s1a = wmma_f16(qa[ks], b1, s1a);
        }

        // Online softmax; row m = qbase + r + 8*hl sits at accumulator reg r,
        // replicated over 16 lanes per half -> 16-lane butterfly reductions.
        f32x8 corrv;
#pragma unroll
        for (int r = 0; r < 8; ++r) {
            const int m = qbase + r + (hl << 3);
            const int n0 = s0 + ln, n1 = n0 + 16;
            float x0 = (n0 <= m) ? s0a[r] : -1e30f;
            float x1 = (n1 <= m) ? s1a[r] : -1e30f;
            float tmax = fmaxf(x0, x1);
#pragma unroll
            for (int off = 1; off < 16; off <<= 1)
                tmax = fmaxf(tmax, __shfl_xor(tmax, off, 32));
            float mnew = fmaxf(mrow[r], tmax);
            float corr = __expf(mrow[r] - mnew);
            mrow[r] = mnew;
            corrv[r] = corr;
            float p0 = __expf(x0 - mnew);
            float p1 = __expf(x1 - mnew);
            float rs = p0 + p1;
#pragma unroll
            for (int off = 1; off < 16; off <<= 1)
                rs += __shfl_xor(rs, off, 32);
            lrow[r] = lrow[r] * corr + rs;
            Pw[(r + (hl << 3)) * 32 + ln]      = (_Float16)p0;
            Pw[(r + (hl << 3)) * 32 + 16 + ln] = (_Float16)p1;
        }
#pragma unroll
        for (int dt = 0; dt < 8; ++dt) o[dt] = o[dt] * corrv;   // v_pk_mul_f32
        __syncthreads();

        // O += P (16x32) * V (32x128); V is [d][s] in LDS so each B-frag is a
        // contiguous v16h (2 x ds_load_b128).
        union { v16h v; unsigned u[8]; } pa;
#pragma unroll
        for (int g = 0; g < 8; ++g)
            pa.u[g] = *(const unsigned*)&Pw[ln * 32 + a_kidx(2 * g, hl)];
#pragma unroll
        for (int dt = 0; dt < 8; ++dt) {
            v16h vb = *(const v16h*)&Vv[(size_t)(dt * 16 + ln) * 32 + (hl << 4)];
            o[dt] = wmma_f16(pa.v, vb, o[dt]);
        }
        __syncthreads();   // buf[cur] consumed; safe to refill at it+2
    }

    // Normalize (8 reciprocals, packed multiplies) and write token-major O.
    f32x8 linv;
#pragma unroll
    for (int r = 0; r < 8; ++r) linv[r] = 1.0f / lrow[r];
    const int b = bh / H_, h = bh - b * H_;
#pragma unroll
    for (int dt = 0; dt < 8; ++dt) {
        f32x8 ov = o[dt] * linv;
#pragma unroll
        for (int r = 0; r < 8; ++r) {
            int t = qbase + r + (hl << 3);
            Oh[((size_t)(b * T_ + t)) * INNER + h * K_ + dt * 16 + ln] =
                (_Float16)ov[r];
        }
    }
}

// ---------------------------------------------------------------------------
// out = O (8192x1024 f16) @ Wu + bu, using transposed f16 WuT [128][1024].
// ---------------------------------------------------------------------------
__global__ void __launch_bounds__(32)
outproj_kernel(const _Float16* __restrict__ Oh, const _Float16* __restrict__ WuT,
               const float* __restrict__ bu, float* __restrict__ out) {
    const int lane = threadIdx.x & 31;
    const int ln = lane & 15, hl = lane >> 4;
    const int Mbase = blockIdx.x * 16;
    const int Nbase = blockIdx.y * 16;
    const int n = Nbase + ln;

    f32x8 acc = {};
    const _Float16* orow = Oh + (size_t)(Mbase + ln) * INNER;
    for (int ks = 0; ks < 32; ++ks) {
        union { v16h v; unsigned u[8]; } a;
#pragma unroll
        for (int g = 0; g < 8; ++g)
            a.u[g] = *(const unsigned*)&orow[ks * 32 + a_kidx(2 * g, hl)];
        v16h b = *(const v16h*)&WuT[(size_t)n * INNER + ks * 32 + (hl << 4)];
        acc = wmma_f16(a.v, b, acc);
    }
    const float bias = bu[n];
#pragma unroll
    for (int r = 0; r < 8; ++r)
        out[(size_t)(Mbase + r + (hl << 3)) * K_ + n] = acc[r] + bias;
}

// ---------------------------------------------------------------------------
extern "C" void kernel_launch(void* const* d_in, const int* in_sizes, int n_in,
                              void* d_out, int out_size, void* d_ws, size_t ws_size,
                              hipStream_t stream) {
    (void)in_sizes; (void)n_in; (void)out_size; (void)ws_size;
    const float* x  = (const float*)d_in[0];
    const float* Wq = (const float*)d_in[1];
    const float* Wk = (const float*)d_in[2];
    const float* Wv = (const float*)d_in[3];
    const float* Wu = (const float*)d_in[4];
    const float* bu = (const float*)d_in[5];
    float* out = (float*)d_out;

    // Workspace layout (f16 halves): Q,K [b][h][t][d]; V [b][h][d][t];
    // O [b*T][1024]; Xh [8192][128]; WqT/WkT/WvT [1024][128]; WuT [128][1024].
    const size_t NE = (size_t)B_ * H_ * T_ * K_;          // 8,388,608
    const size_t NX = (size_t)B_ * T_ * K_;               // 1,048,576
    const size_t NW = (size_t)K_ * INNER;                 //   131,072
    _Float16* Qh  = (_Float16*)d_ws;
    _Float16* Kh  = Qh + NE;
    _Float16* Vh  = Kh + NE;
    _Float16* Oh  = Vh + NE;
    _Float16* Xh  = Oh + NE;
    _Float16* WqT = Xh + NX;
    _Float16* WkT = WqT + NW;
    _Float16* WvT = WkT + NW;
    _Float16* WuT = WvT + NW;

    const float qscale = 0.08838834764831845f;   // 128^-0.5 (softmax scale)
    const float vscale = 0.29730177875068026f;   // 128^-0.25 (reference V scale)

    cvt_x_kernel<<<dim3((unsigned)(NX / 2 / 256)), 256, 0, stream>>>(x, Xh);
    cvt_wT_kernel<<<dim3(INNER), 128, 0, stream>>>(Wq, WqT, K_, INNER, qscale);
    cvt_wT_kernel<<<dim3(INNER), 128, 0, stream>>>(Wk, WkT, K_, INNER, 1.0f);
    cvt_wT_kernel<<<dim3(INNER), 128, 0, stream>>>(Wv, WvT, K_, INNER, vscale);
    cvt_wT_kernel<<<dim3(K_), 256, 0, stream>>>(Wu, WuT, INNER, K_, 1.0f);

    dim3 ggrid(B_ * T_ / 16, INNER / 16);        // 512 x 64
    qkv_kernel<<<ggrid, 32, 0, stream>>>(Xh, WqT, WkT, WvT, Qh, Kh, Vh);

    dim3 agrid(T_ / 64, B_ * H_);                // 32 x 32, 4 waves per block
    attn_kernel<<<agrid, 128, 0, stream>>>(Qh, Kh, Vh, Oh);

    dim3 ogrid(B_ * T_ / 16, K_ / 16);           // 512 x 8
    outproj_kernel<<<ogrid, 32, 0, stream>>>(Oh, WuT, bu, out);
}
